// RNNTJointer_17059610100287
// MI455X (gfx1250) — matrix operand order repaired and checked by reading, not compiled, not run on hardware
//
#include <hip/hip_runtime.h>
#include <math.h>

typedef __attribute__((ext_vector_type(16))) _Float16 v16h;
typedef __attribute__((ext_vector_type(8)))  _Float16 v8h;
typedef __attribute__((ext_vector_type(4)))  _Float16 v4h;
typedef __attribute__((ext_vector_type(8)))  float    v8f;

#define WMMA_F16(a, b, c) \
  __builtin_amdgcn_wmma_f32_16x16x32_f16(false, (a), false, (b), (short)0, (c), false, false)

namespace {
constexpr int Bb = 4, T = 256, U = 128, E = 512, D = 640, H = 512, V = 640;
constexpr int UB = 64;       // u-rows per workgroup (4 M-tiles) -> 4x B-fragment reuse
constexpr int HP = H + 8;    // padded LDS row stride (halves): rows land on distinct bank groups
}

// Fast saturating tanh: 1 - 2/(e^{2x}+1).  exp overflow -> +1, underflow -> -1 (no NaN).
// One v_exp_f32 + one v_rcp_f32; error ~1e-6 << f16 quantization of the A-tile.
__device__ __forceinline__ float fast_tanh(float x) {
  float ex = __expf(2.0f * x);
  return 1.0f - __fdividef(2.0f, ex + 1.0f);
}

// Assemble a 16x32 f16 A fragment (per-lane 16 halves) from two 16B chunks.
__device__ __forceinline__ v16h load_a_frag(const _Float16* p) {
  v8h lo = *(const v8h*)(p);        // K {0..7}  (lanes 0-15) / {8..15}  (lanes 16-31)
  v8h hi = *(const v8h*)(p + 16);   // K {16..23}            / {24..31}
  return __builtin_shufflevector(lo, hi, 0,1,2,3,4,5,6,7,8,9,10,11,12,13,14,15);
}

// ---------------- fp32 -> fp16 convert (vectorized x4) ----------------
__global__ void cvt_f16(const float* __restrict__ src, _Float16* __restrict__ dst, int n4) {
  int i = blockIdx.x * 256 + threadIdx.x;
  if (i >= n4) return;
  float4 f = ((const float4*)src)[i];
  v4h o;
  o[0] = (_Float16)f.x; o[1] = (_Float16)f.y; o[2] = (_Float16)f.z; o[3] = (_Float16)f.w;
  ((v4h*)dst)[i] = o;
}

// ------- fp32 [K][N] -> fp16 [N][K] transpose (LDS-tiled, coalesced both ways) -------
__global__ void cvtT_f16(const float* __restrict__ src, _Float16* __restrict__ dst,
                         int K, int N) {
  __shared__ float tile[16][17];
  int n0 = blockIdx.x * 16, k0 = blockIdx.y * 16;
  int tx = threadIdx.x, ty = threadIdx.y;
  tile[ty][tx] = src[(size_t)(k0 + ty) * N + (n0 + tx)];
  __syncthreads();
  dst[(size_t)(n0 + ty) * K + (k0 + tx)] = (_Float16)tile[tx][ty];
}

// ---------------- WMMA projection GEMM: C_f16[M][N] = A_f16[M][K] @ Bt_f16[N][K] + bias ----------------
__global__ __launch_bounds__(256) void proj_wmma(
    const _Float16* __restrict__ A, const _Float16* __restrict__ Bt,
    const float* __restrict__ bias, _Float16* __restrict__ C,
    int M, int K, int N) {
  int wave = threadIdx.x >> 5;
  int lane = threadIdx.x & 31;
  int tile = blockIdx.x * 8 + wave;
  int ntiles = N >> 4;
  int mt = tile / ntiles, nt = tile % ntiles;
  int m0 = mt * 16, n0 = nt * 16;
  if (m0 >= M) return;

  int l15 = lane & 15, lh = lane >> 4;
  const _Float16* arow = A + (size_t)(m0 + l15) * K + lh * 8;
  const _Float16* brow = Bt + (size_t)(n0 + l15) * K + lh * 8;

  int vcol = n0 + l15;
  float bv = bias ? bias[vcol] : 0.0f;
  v8f acc;
  #pragma unroll
  for (int r = 0; r < 8; ++r) acc[r] = bv;   // bias folded into C

  for (int k0 = 0; k0 < K; k0 += 32) {
    v16h a = load_a_frag(arow + k0);
    v16h b = load_a_frag(brow + k0);
    acc = WMMA_F16(a, b, acc);
  }
  int mb = m0 + lh * 8;   // C/D layout: VGPR r -> M = r (+8 for lanes 16-31)
  #pragma unroll
  for (int r = 0; r < 8; ++r)
    C[(size_t)(mb + r) * N + vcol] = (_Float16)acc[r];
}

// ---------------- Joint kernel: out[b,t,u,v] = tanh(enc+dec) @ WoutT + b_out ----------------
// One WG = (b, t, 64 u-rows). 4 M-tiles register-blocked per wave: each B fragment
// fetched from (L2-resident) WoutT feeds 4 WMMAs -> 4x less L2 traffic.
__global__ __launch_bounds__(256) void joint_wmma(
    const _Float16* __restrict__ enc_h,   // [B*T][H] f16
    const _Float16* __restrict__ dec_h,   // [B*U][H] f16
    const _Float16* __restrict__ WoutT,   // [V][H]  f16 (transposed)
    const float*    __restrict__ b_out,   // [V]
    float*          __restrict__ out) {   // [B,T,U,V] f32
  extern __shared__ _Float16 As[];        // UB x HP = 64 x 520 f16 = 66,560 B (dynamic)

  int wg   = blockIdx.x;
  int ublk = wg & 1;                      // U/UB = 2
  int t    = (wg >> 1) & (T - 1);
  int b    = wg >> 9;

  const _Float16* encRow  = enc_h + (size_t)(b * T + t) * H;
  const _Float16* decBase = dec_h + (size_t)(b * U + ublk * UB) * H;
  int tid = threadIdx.x;

  // Phase 1: A-tile = f16(tanh(enc_row + dec_rows)) into LDS, 8 halves per step
  for (int idx = tid; idx < UB * (H / 8); idx += 256) {
    int u  = idx >> 6;                    // H/8 = 64 chunks per row
    int k8 = (idx & 63) << 3;
    v8h e = *(const v8h*)(encRow + k8);
    v8h d = *(const v8h*)(decBase + (size_t)u * H + k8);
    v8h r;
    #pragma unroll
    for (int i = 0; i < 8; ++i)
      r[i] = (_Float16)fast_tanh((float)e[i] + (float)d[i]);
    *(v8h*)(&As[(size_t)u * HP + k8]) = r;
  }
  __syncthreads();

  // Phase 2: each wave owns 5 distinct N-tiles; per N-tile accumulate 4 M-tiles.
  int wave = tid >> 5, lane = tid & 31;
  int l15 = lane & 15, lh = lane >> 4;
  const _Float16* a0 = &As[(size_t)( 0 + l15) * HP + lh * 8];
  const _Float16* a1 = &As[(size_t)(16 + l15) * HP + lh * 8];
  const _Float16* a2 = &As[(size_t)(32 + l15) * HP + lh * 8];
  const _Float16* a3 = &As[(size_t)(48 + l15) * HP + lh * 8];
  size_t outRow0 = ((size_t)(b * T + t) * U + ublk * UB + lh * 8);

  for (int nt = wave; nt < V / 16; nt += 8) {
    int vcol = nt * 16 + l15;
    float bv = b_out[vcol];
    v8f acc0, acc1, acc2, acc3;
    #pragma unroll
    for (int r = 0; r < 8; ++r) { acc0[r] = bv; acc1[r] = bv; acc2[r] = bv; acc3[r] = bv; }

    const _Float16* brow = WoutT + (size_t)(nt * 16 + l15) * H + lh * 8;
    #pragma unroll 4
    for (int k0 = 0; k0 < H; k0 += 32) {
      v16h bf = load_a_frag(brow + k0);   // 2x global_load_b128, reused 4x
      acc0 = WMMA_F16(load_a_frag(a0 + k0), bf, acc0);
      acc1 = WMMA_F16(load_a_frag(a1 + k0), bf, acc1);
      acc2 = WMMA_F16(load_a_frag(a2 + k0), bf, acc2);
      acc3 = WMMA_F16(load_a_frag(a3 + k0), bf, acc3);
    }
    float* obase = out + outRow0 * V + vcol;
    #pragma unroll
    for (int r = 0; r < 8; ++r) {
      __builtin_nontemporal_store(acc0[r], obase + (size_t)( 0 + r) * V);
      __builtin_nontemporal_store(acc1[r], obase + (size_t)(16 + r) * V);
      __builtin_nontemporal_store(acc2[r], obase + (size_t)(32 + r) * V);
      __builtin_nontemporal_store(acc3[r], obase + (size_t)(48 + r) * V);
    }
  }
}

extern "C" void kernel_launch(void* const* d_in, const int* in_sizes, int n_in,
                              void* d_out, int out_size, void* d_ws, size_t ws_size,
                              hipStream_t stream) {
  const float* enc_out = (const float*)d_in[0];   // [4,256,512]
  const float* dec_out = (const float*)d_in[1];   // [4,128,640]
  const float* W_enc   = (const float*)d_in[2];   // [512,512]
  const float* b_enc   = (const float*)d_in[3];   // [512]
  const float* W_dec   = (const float*)d_in[4];   // [640,512]
  const float* W_out   = (const float*)d_in[5];   // [512,640]
  const float* b_out   = (const float*)d_in[6];   // [640]
  float* out = (float*)d_out;

  char* ws = (char*)d_ws;
  _Float16* encA  = (_Float16*)(ws + 0);          // enc_out f16   1,048,576 B
  _Float16* decA  = (_Float16*)(ws + 1048576);    // dec_out f16     655,360 B
  _Float16* WencT = (_Float16*)(ws + 1703936);    // [H][E] f16      524,288 B
  _Float16* WdecT = (_Float16*)(ws + 2228224);    // [H][D] f16      655,360 B
  _Float16* WoutT = (_Float16*)(ws + 2883584);    // [V][H] f16      655,360 B
  _Float16* enc_h = (_Float16*)(ws + 3538944);    // [B*T][H] f16  1,048,576 B
  _Float16* dec_h = (_Float16*)(ws + 4587520);    // [B*U][H] f16    524,288 B
  // total ws use: 5,111,808 B

  { int n4 = Bb * T * E / 4; cvt_f16<<<(n4 + 255) / 256, 256, 0, stream>>>(enc_out, encA, n4); }
  { int n4 = Bb * U * D / 4; cvt_f16<<<(n4 + 255) / 256, 256, 0, stream>>>(dec_out, decA, n4); }
  cvtT_f16<<<dim3(H / 16, E / 16), dim3(16, 16), 0, stream>>>(W_enc, WencT, E, H);
  cvtT_f16<<<dim3(H / 16, D / 16), dim3(16, 16), 0, stream>>>(W_dec, WdecT, D, H);
  cvtT_f16<<<dim3(V / 16, H / 16), dim3(16, 16), 0, stream>>>(W_out, WoutT, H, V);

  proj_wmma<<<(Bb * T / 16) * (H / 16) / 8, 256, 0, stream>>>(encA, WencT, b_enc, enc_h, Bb * T, E, H);
  proj_wmma<<<(Bb * U / 16) * (H / 16) / 8, 256, 0, stream>>>(decA, WdecT, nullptr, dec_h, Bb * U, D, H);

  joint_wmma<<<Bb * T * (U / UB), 256, UB * HP * sizeof(_Float16), stream>>>(
      enc_h, dec_h, WoutT, b_out, out);
}
